// Quantizer_29789893165324
// MI455X (gfx1250) — matrix operand order repaired and checked by reading, not compiled, not run on hardware
//
#include <hip/hip_runtime.h>
#include <hip/hip_bf16.h>

#define N_ROWS 65536
#define DIM    256
#define K_CODES 1024
#define TILE_N 16
#define PITCH  264   // ushorts per LDS row (528B) -> conflict-free b128 fragment reads

typedef __attribute__((ext_vector_type(16))) __bf16 v16bf;
typedef __attribute__((ext_vector_type(8)))  float  v8f;

union Frag { v16bf v; unsigned short s[16]; uint4 q[2]; };

__device__ __forceinline__ unsigned short f2bf(float f){
    unsigned u = __float_as_uint(f);
    unsigned r = ((u >> 16) & 1u) + 0x7fffu;   // round-to-nearest-even
    return (unsigned short)((u + r) >> 16);
}
__device__ __forceinline__ float bf2f(unsigned short h){
    return __uint_as_float(((unsigned)h) << 16);
}

// --- Kernel 1: codebook -> bf16 hi/lo split, ||e||^2, zero histogram ---------
__global__ void __launch_bounds__(256) vq_prep(const float* __restrict__ cb,
                                               unsigned short* __restrict__ cb_hi,
                                               unsigned short* __restrict__ cb_lo,
                                               float* __restrict__ enorm,
                                               int* __restrict__ counts){
    const int lane = threadIdx.x & 31;
    const int wave = threadIdx.x >> 5;
    const int code = blockIdx.x * 8 + wave;          // grid=128 -> 1024 codes
    if (blockIdx.x < 4) counts[blockIdx.x * 256 + threadIdx.x] = 0;

    const float* src = cb + (size_t)code * DIM + lane * 8;
    float4 f0 = ((const float4*)src)[0];
    float4 f1 = ((const float4*)src)[1];
    float fv[8] = {f0.x,f0.y,f0.z,f0.w, f1.x,f1.y,f1.z,f1.w};

    union { unsigned short s[8]; uint4 q; } hi, lo;
    float ss = 0.0f;
#pragma unroll
    for (int i = 0; i < 8; ++i){
        float f = fv[i];
        ss += f * f;
        unsigned short h = f2bf(f);
        hi.s[i] = h;
        lo.s[i] = f2bf(f - bf2f(h));
    }
    ((uint4*)(cb_hi + (size_t)code * DIM + lane * 8))[0] = hi.q;
    ((uint4*)(cb_lo + (size_t)code * DIM + lane * 8))[0] = lo.q;

#pragma unroll
    for (int off = 16; off >= 1; off >>= 1) ss += __shfl_xor(ss, off, 32);
    if (lane == 0) enorm[code] = ss;
}

// --- Kernel 2: fused distance GEMM (bf16x3 WMMA) + per-row argmin ------------
__global__ void __launch_bounds__(256) vq_argmin(const float* __restrict__ x,
                                                 const unsigned short* __restrict__ cb_hi,
                                                 const unsigned short* __restrict__ cb_lo,
                                                 const float* __restrict__ enorm,
                                                 int* __restrict__ out_idx_i,
                                                 float* __restrict__ out_idx_f){
    __shared__ unsigned short lds_hi[TILE_N * PITCH];
    __shared__ unsigned short lds_lo[TILE_N * PITCH];

    const int tid  = threadIdx.x;
    const int lane = tid & 31;
    const int wave = tid >> 5;
    const int rowbase = blockIdx.x * 128 + wave * 16;   // grid=512 -> 65536 rows
    const int row  = rowbase + (lane & 15);
    const int koff = (lane < 16) ? 0 : 8;               // ISA 16-bit A-matrix layout

    // Build resident A fragments: 16x256 row tile, bf16 hi/lo split.
    Frag a_hi[8], a_lo[8];
    const float* rowp = x + (size_t)row * DIM;
#pragma unroll
    for (int ci = 0; ci < 8; ++ci){
        const int k0 = ci * 32;
        float4 f0 = *(const float4*)(rowp + k0 + koff);
        float4 f1 = *(const float4*)(rowp + k0 + koff + 4);
        float4 f2 = *(const float4*)(rowp + k0 + 16 + koff);
        float4 f3 = *(const float4*)(rowp + k0 + 16 + koff + 4);
        float fv[16] = {f0.x,f0.y,f0.z,f0.w, f1.x,f1.y,f1.z,f1.w,
                        f2.x,f2.y,f2.z,f2.w, f3.x,f3.y,f3.z,f3.w};
#pragma unroll
        for (int e = 0; e < 16; ++e){
            unsigned short h = f2bf(fv[e]);
            a_hi[ci].s[e] = h;
            a_lo[ci].s[e] = f2bf(fv[e] - bf2f(h));
        }
    }

    float best[8];
    int   bidx[8];
#pragma unroll
    for (int j = 0; j < 8; ++j){ best[j] = 3.4e38f; bidx[j] = 0; }

    const int scode = tid >> 4;          // staging: code row within tile
    const int skb   = (tid & 15) * 16;   // staging: K base (16 bf16 = 32B per thread)

    for (int n0 = 0; n0 < K_CODES; n0 += TILE_N){
        // Cooperative stage of 16-code B tile (hi + lo) into padded LDS.
        {
            const size_t g = (size_t)(n0 + scode) * DIM + skb;
            const uint4* ph = (const uint4*)(cb_hi + g);
            const uint4* pl = (const uint4*)(cb_lo + g);
            uint4* dh = (uint4*)(lds_hi + scode * PITCH + skb);
            uint4* dl = (uint4*)(lds_lo + scode * PITCH + skb);
            dh[0] = ph[0]; dh[1] = ph[1];
            dl[0] = pl[0]; dl[1] = pl[1];
            if (n0 + TILE_N < K_CODES){
                __builtin_prefetch(cb_hi + g + TILE_N * DIM, 0, 0);
                __builtin_prefetch(cb_lo + g + TILE_N * DIM, 0, 0);
            }
        }
        __syncthreads();

        // 24 chained v_wmma_f32_16x16x32_bf16: dot = Ahi*Bhi + Ahi*Blo + Alo*Bhi
        v8f c = {};
        const int ksel = (lane < 16) ? 0 : 8;   // B layout: lanes 0-15 K0-7,16-23; 16-31 K8-15,24-31
        const int crow = (lane & 15) * PITCH;
#pragma unroll
        for (int ci = 0; ci < 8; ++ci){
            Frag bh, bl;
            const uint4* ph = (const uint4*)(lds_hi + crow + ci * 32 + ksel);
            const uint4* pl = (const uint4*)(lds_lo + crow + ci * 32 + ksel);
            bh.q[0] = ph[0]; bh.q[1] = ph[2];
            bl.q[0] = pl[0]; bl.q[1] = pl[2];
            c = __builtin_amdgcn_wmma_f32_16x16x32_bf16(false, a_hi[ci].v, false, bh.v, (short)0, c, false, false);
            c = __builtin_amdgcn_wmma_f32_16x16x32_bf16(false, a_hi[ci].v, false, bl.v, (short)0, c, false, false);
            c = __builtin_amdgcn_wmma_f32_16x16x32_bf16(false, a_lo[ci].v, false, bh.v, (short)0, c, false, false);
        }

        // score = ||e||^2 - 2 x.e  (||x||^2 is argmin-invariant)
        const int   n  = n0 + (lane & 15);
        const float en = enorm[n];
#pragma unroll
        for (int j = 0; j < 8; ++j){
            float s = en - 2.0f * c[j];
            if (s < best[j]) { best[j] = s; bidx[j] = n; }
        }
        __syncthreads();
    }

    // Min+index reduction over the 16 lanes of each half-wave (first-index ties).
#pragma unroll
    for (int j = 0; j < 8; ++j){
        float s = best[j]; int i = bidx[j];
#pragma unroll
        for (int off = 8; off >= 1; off >>= 1){
            float os = __shfl_xor(s, off, 32);
            int   oi = __shfl_xor(i, off, 32);
            if (os < s || (os == s && oi < i)) { s = os; i = oi; }
        }
        if ((lane & 15) == 0){
            // C/D layout: VGPR j holds row M=j (lanes 0-15) or M=j+8 (lanes 16-31)
            const int r = rowbase + j + ((lane >> 4) << 3);
            out_idx_i[r] = i;
            out_idx_f[r] = (float)i;
        }
    }
}

// --- Kernel 3: gather quantized rows (straight-through arith) + histogram ----
__global__ void __launch_bounds__(256) vq_gather(const float* __restrict__ x,
                                                 const float* __restrict__ cb,
                                                 const int* __restrict__ idx,
                                                 float* __restrict__ outq,
                                                 int* __restrict__ counts){
    const int lane = threadIdx.x & 31;
    const int wave = threadIdx.x >> 5;
    const int row  = blockIdx.x * 8 + wave;        // grid=8192 -> 65536 rows
    const int k = idx[row];
    const float4* q = (const float4*)(cb   + (size_t)k   * DIM + lane * 8);
    const float4* f = (const float4*)(x    + (size_t)row * DIM + lane * 8);
    float4*       d = (float4*)      (outq + (size_t)row * DIM + lane * 8);
    float4 q0 = q[0], q1 = q[1], f0 = f[0], f1 = f[1];
    float4 o0, o1;
    o0.x = f0.x + (q0.x - f0.x); o0.y = f0.y + (q0.y - f0.y);
    o0.z = f0.z + (q0.z - f0.z); o0.w = f0.w + (q0.w - f0.w);
    o1.x = f1.x + (q1.x - f1.x); o1.y = f1.y + (q1.y - f1.y);
    o1.z = f1.z + (q1.z - f1.z); o1.w = f1.w + (q1.w - f1.w);
    d[0] = o0; d[1] = o1;
    if (lane == 0) atomicAdd(&counts[k], 1);
}

// --- Kernel 4: perplexity ----------------------------------------------------
__global__ void __launch_bounds__(1024) vq_ppl(const int* __restrict__ counts,
                                               float* __restrict__ out){
    __shared__ float red[1024];
    const int t = threadIdx.x;
    const float p = (float)counts[t] * (1.0f / (float)N_ROWS);
    red[t] = -p * logf(p + 1e-10f);
    __syncthreads();
    for (int s = 512; s > 0; s >>= 1){
        if (t < s) red[t] += red[t + s];
        __syncthreads();
    }
    if (t == 0) out[0] = expf(red[0]);
}

extern "C" void kernel_launch(void* const* d_in, const int* in_sizes, int n_in,
                              void* d_out, int out_size, void* d_ws, size_t ws_size,
                              hipStream_t stream){
    (void)in_sizes; (void)n_in; (void)out_size; (void)ws_size;
    const float* x  = (const float*)d_in[0];   // [65536, 256]
    const float* cb = (const float*)d_in[1];   // [1024, 256]

    float* outq = (float*)d_out;                       // quantized: 16777216
    float* outp = outq + (size_t)N_ROWS * DIM;         // perplexity: 1
    float* outi = outp + 1;                            // indices:    65536

    char* ws = (char*)d_ws;
    unsigned short* cb_hi = (unsigned short*)ws;                    // 512 KB
    unsigned short* cb_lo = (unsigned short*)(ws + 524288);         // 512 KB
    float* enorm = (float*)(ws + 1048576);                          //   4 KB
    int*   counts = (int*)(ws + 1048576 + 4096);                    //   4 KB
    int*   idxbuf = (int*)(ws + 1048576 + 8192);                    // 256 KB

    vq_prep  <<<128,  256, 0, stream>>>(cb, cb_hi, cb_lo, enorm, counts);
    vq_argmin<<<512,  256, 0, stream>>>(x, cb_hi, cb_lo, enorm, idxbuf, outi);
    vq_gather<<<8192, 256, 0, stream>>>(x, cb, idxbuf, outq, counts);
    vq_ppl   <<<1,   1024, 0, stream>>>(counts, outp);
}
